// DependencyParseModel_53644141527732
// MI455X (gfx1250) — matrix operand order repaired and checked by reading, not compiled, not run on hardware
//
#include <hip/hip_runtime.h>
#include <hip/hip_bf16.h>
#include <math.h>

// ---------------- model dims ----------------
#define TT    256   // sentence length
#define INQ   400   // lstm input (word 300 + tag 100)
#define INP   416   // padded to multiple of 32 for WMMA K-loop
#define HH    400   // hidden
#define G4    1600  // 4*H gates
#define BIc   800   // bilstm output
#define MLP   1600  // pairwise MLP width
#define LBL   40    // label count
#define GSCAN 8     // workgroups cooperating per LSTM direction
#define JB    (HH / GSCAN)   // 50 hidden units owned per workgroup
#define NT    4     // n-subtiles per wave in GEMM (16x64 output per wave)

typedef __attribute__((ext_vector_type(16))) _Float16 v16h;
typedef __attribute__((ext_vector_type(8)))  float    v8f;

// ---------------- small utility kernels ----------------
__global__ void zero_i32(int* p, int n) {
    int i = blockIdx.x * blockDim.x + threadIdx.x;
    if (i < n) p[i] = 0;
}

// f32 [rows, cols] -> f16 [rows, dld] (zero-padded columns cols..dld)
__global__ void convert2d(const float* __restrict__ src, int cols,
                          _Float16* __restrict__ dst, int dld, long total) {
    long i = (long)blockIdx.x * blockDim.x + threadIdx.x;
    if (i >= total) return;
    int c = (int)(i % dld);
    long r = i / dld;
    dst[i] = (c < cols) ? (_Float16)src[r * (long)cols + c] : (_Float16)0.f;
}

// gather word/tag embeddings, concat, pad to INP, emit f16
__global__ void embed_concat(const int* __restrict__ words, const int* __restrict__ tags,
                             const float* __restrict__ ew, const float* __restrict__ et,
                             _Float16* __restrict__ xh) {
    int i = blockIdx.x * blockDim.x + threadIdx.x;
    if (i >= TT * INP) return;
    int t = i / INP, c = i % INP;
    float v = 0.f;
    if (c < 300)       v = ew[(size_t)words[t] * 300 + c];
    else if (c < 400)  v = et[(size_t)tags[t] * 100 + (c - 300)];
    xh[i] = (_Float16)v;
}

// load initial h (f16, parity-0 buffer) and c (f32) for one BiLSTM layer
__global__ void lstm_init(const float* __restrict__ h0, const float* __restrict__ c0,
                          int layer, _Float16* __restrict__ hbuf, float* __restrict__ cst) {
    int i = blockIdx.x * blockDim.x + threadIdx.x;
    if (i >= 2 * HH) return;
    int d = i / HH, j = i % HH;
    hbuf[(0 * 2 + d) * HH + j] = (_Float16)h0[(2 * layer + d) * HH + j];
    cst[i] = c0[(2 * layer + d) * HH + j];
}

// ---------------- WMMA GEMM:  C[M,N] = act( A[M,K] @ W[N,K]^T + bias + bias2 + Cinit ) ----------------
// A,W f16; C f32 accum. One wave computes a 16x64 output tile (NT=4 n-subtiles),
// reusing one A fragment across 4 WMMAs per K-step. Fragment layouts per CDNA5 ISA 7.12.2:
//   A 16x32: lane<16 holds K {0..7,16..23}; lane>=16 holds K {8..15,24..31} of row (lane&15)
//   B 32x16: lane<16 holds K {0..7,16..23}; lane>=16 holds K {8..15,24..31} of column n  (== row n of W)
//   C/D:     N = lane&15, M = (lane>>4)*8 + vgpr
__global__ void gemm_wmma(const _Float16* __restrict__ A, int lda, const int* __restrict__ rowidx,
                          const _Float16* __restrict__ W, int ldw,
                          const float* __restrict__ bias, const float* __restrict__ bias2,
                          const float* __restrict__ Cinit, float* __restrict__ Cout,
                          _Float16* __restrict__ CoutH,
                          int M, int N, int K, int act) {
    int lane = threadIdx.x & 31;
    int r16  = lane & 15;
    int hi   = lane >> 4;
    int m0 = blockIdx.y * 16, n0 = blockIdx.x * (16 * NT);

    int rowA = m0 + r16;
    if (rowidx) rowA = rowidx[rowA];          // fused gather (hv[arcs])
    const _Float16* pa_base = A + (size_t)rowA * lda + hi * 8;

    const _Float16* pb_base[NT];
#pragma unroll
    for (int s = 0; s < NT; ++s) {
        int rowW = n0 + s * 16 + r16;
        int rowWc = (rowW < N) ? rowW : 0;    // clamp; stores are guarded
        pb_base[s] = W + (size_t)rowWc * ldw + hi * 8;
    }

    v8f c[NT];
#pragma unroll
    for (int s = 0; s < NT; ++s) c[s] = (v8f){0.f, 0.f, 0.f, 0.f, 0.f, 0.f, 0.f, 0.f};

    for (int kk = 0; kk < K; kk += 32) {
        v16h a;
        const _Float16* pa = pa_base + kk;
#pragma unroll
        for (int i = 0; i < 8; ++i) { a[i] = pa[i]; a[8 + i] = pa[16 + i]; }
        v16h b[NT];
#pragma unroll
        for (int s = 0; s < NT; ++s) {
            const _Float16* pb = pb_base[s] + kk;
#pragma unroll
            for (int i = 0; i < 8; ++i) { b[s][i] = pb[i]; b[s][8 + i] = pb[16 + i]; }
        }
#pragma unroll
        for (int s = 0; s < NT; ++s)
            c[s] = __builtin_amdgcn_wmma_f32_16x16x32_f16(false, a, false, b[s], (short)0, c[s],
                                                          false, false);
    }

    int mb = m0 + hi * 8;
#pragma unroll
    for (int s = 0; s < NT; ++s) {
        int n = n0 + s * 16 + r16;
        if (n < N) {
#pragma unroll
            for (int r = 0; r < 8; ++r) {
                float v = c[s][r];
                size_t o = (size_t)(mb + r) * N + n;
                if (Cinit) v += Cinit[o];
                if (bias)  v += bias[n];
                if (bias2) v += bias2[n];
                if (act == 1) v = tanhf(v);
                if (Cout)  Cout[o] = v;
                if (CoutH) CoutH[o] = (_Float16)v;
            }
        }
    }
}

// ---------------- cooperative LSTM recurrence ----------------
// grid = (GSCAN, 2 dirs). Each block owns gate rows {q*400 + [jb, jb+JB)} for q=0..3,
// so its c/h slice [jb, jb+JB) is fully local. h vector double-buffered in global
// (L2 resident); per-step barrier via atomic counters (one per dir per step).
__global__ void lstm_scan(const float* __restrict__ XW,      // [2][T][G4] x@Wih^T + bih + bhh
                          const _Float16* __restrict__ WhhH, // [2][G4][HH]
                          _Float16* __restrict__ hbuf,       // [2 parity][2 dir][HH]
                          const float* __restrict__ cst0,    // [2][HH] initial c
                          _Float16* __restrict__ hs_out,     // [T][2*HH], col d*HH
                          int* __restrict__ cnt)             // [2][T]
{
    int d = blockIdx.y, b = blockIdx.x, tid = threadIdx.x;
    int jb = b * JB;
    __shared__ _Float16 sh[HH];
    __shared__ float    sg[4 * JB];

    float creg = (tid < JB) ? cst0[d * HH + jb + tid] : 0.f;
    const _Float16* Whh = WhhH + (size_t)d * G4 * HH;
    const float*    XWd = XW   + (size_t)d * TT * G4;
    int* cn = cnt + d * TT;

    for (int s = 0; s < TT; ++s) {
        int t   = d ? (TT - 1 - s) : s;
        int par = s & 1;
        __syncthreads();
        for (int i = tid; i < HH; i += blockDim.x)
            sh[i] = hbuf[(par * 2 + d) * HH + i];
        __syncthreads();

        if (tid < 4 * JB) {                     // one gate row per thread
            int q = tid / JB, jj = tid % JB;
            int row = q * HH + jb + jj;
            const _Float16* w = Whh + (size_t)row * HH;
            float acc = 0.f;
            for (int k = 0; k < HH; ++k) acc += (float)w[k] * (float)sh[k];
            sg[tid] = acc + XWd[(size_t)t * G4 + row];
        }
        __syncthreads();

        if (tid < JB) {                         // gate fusion for owned hidden units
            float gi = 1.f / (1.f + expf(-sg[tid]));
            float gf = 1.f / (1.f + expf(-sg[JB + tid]));
            float gg = tanhf(sg[2 * JB + tid]);
            float go = 1.f / (1.f + expf(-sg[3 * JB + tid]));
            creg = gf * creg + gi * gg;
            float h = go * tanhf(creg);
            _Float16 hh = (_Float16)h;
            int j = jb + tid;
            hbuf[((par ^ 1) * 2 + d) * HH + j]  = hh;
            hs_out[(size_t)t * (2 * HH) + d * HH + j] = hh;
        }

        // inter-WGP barrier for this direction's step
        __syncthreads();
        __threadfence();
        if (tid == 0) {
            atomicAdd(&cn[s], 1);
            while (__hip_atomic_load(&cn[s], __ATOMIC_ACQUIRE, __HIP_MEMORY_SCOPE_AGENT) < GSCAN) {}
        }
        __syncthreads();
        __threadfence();
    }
}

// ---------------- pairwise arc scorer ----------------
// scores[i][j] = sum_k tanh(Ah[i,k] + Ad[j,k] + ab1[k]) * aW2[k] + ab2 ; diag fixups fused
__global__ void pairwise_arc(const float* __restrict__ Ah, const float* __restrict__ Ad,
                             const float* __restrict__ ab1, const float* __restrict__ aW2,
                             const float* __restrict__ ab2, float* __restrict__ out) {
    __shared__ float sAh[16][65], sAd[16][65], sb1[64], sw2[64];  // pad -> no bank conflicts
    int tx = threadIdx.x, ty = threadIdx.y;
    int lid = ty * 16 + tx;
    int i0 = blockIdx.y * 16, j0 = blockIdx.x * 16;
    float acc = 0.f;
    for (int kk = 0; kk < MLP; kk += 64) {
        __syncthreads();
        for (int e = lid; e < 16 * 64; e += 256) {
            int r = e >> 6, c = e & 63;
            sAh[r][c] = Ah[(size_t)(i0 + r) * MLP + kk + c];
            sAd[r][c] = Ad[(size_t)(j0 + r) * MLP + kk + c];
        }
        if (lid < 64) { sb1[lid] = ab1[kk + lid]; sw2[lid] = aW2[kk + lid]; }
        __syncthreads();
#pragma unroll 4
        for (int k = 0; k < 64; ++k)
            acc += tanhf(sAh[ty][k] + sAd[tx][k] + sb1[k]) * sw2[k];
    }
    int i = i0 + ty, j = j0 + tx;
    float sres = acc + ab2[0];
    if (i == j) sres = 0.f;
    if (i == 0 && j == 0) sres = 1.f;
    out[(size_t)i * TT + j] = sres;
}

// ---------------- host orchestration ----------------
extern "C" void kernel_launch(void* const* d_in, const int* in_sizes, int n_in,
                              void* d_out, int out_size, void* d_ws, size_t ws_size,
                              hipStream_t stream) {
    (void)in_sizes; (void)n_in; (void)out_size; (void)ws_size;
    const int*   words = (const int*)d_in[0];
    const int*   tags  = (const int*)d_in[1];
    const int*   arcs  = (const int*)d_in[2];
    const float* emb_w = (const float*)d_in[3];
    const float* emb_t = (const float*)d_in[4];
    const float* Wih0  = (const float*)d_in[5];
    const float* Whh0  = (const float*)d_in[6];
    const float* bih0  = (const float*)d_in[7];
    const float* bhh0  = (const float*)d_in[8];
    const float* Wih1  = (const float*)d_in[9];
    const float* Whh1  = (const float*)d_in[10];
    const float* bih1  = (const float*)d_in[11];
    const float* bhh1  = (const float*)d_in[12];
    const float* h0    = (const float*)d_in[13];
    const float* c0    = (const float*)d_in[14];
    const float* aW1   = (const float*)d_in[15];
    const float* ab1   = (const float*)d_in[16];
    const float* aW2   = (const float*)d_in[17];
    const float* ab2   = (const float*)d_in[18];
    const float* lW1   = (const float*)d_in[19];
    const float* lb1   = (const float*)d_in[20];
    const float* lW2   = (const float*)d_in[21];
    const float* lb2   = (const float*)d_in[22];
    float* out = (float*)d_out;

    // workspace carve-up (256B aligned regions; ~34 MB total)
    char* wsb = (char*)d_ws;
    size_t off = 0;
    auto take = [&](size_t bytes) -> char* {
        char* p = wsb + off;
        off += (bytes + 255) & ~(size_t)255;
        return p;
    };
    _Float16* wih0h = (_Float16*)take((size_t)2 * G4 * INP * 2);
    _Float16* whh0h = (_Float16*)take((size_t)2 * G4 * HH  * 2);
    _Float16* wih1h = (_Float16*)take((size_t)2 * G4 * BIc * 2);
    _Float16* whh1h = (_Float16*)take((size_t)2 * G4 * HH  * 2);
    _Float16* aW1h  = (_Float16*)take((size_t)MLP * MLP * 2);
    _Float16* lW1h  = (_Float16*)take((size_t)MLP * MLP * 2);
    _Float16* lW2h  = (_Float16*)take((size_t)LBL * MLP * 2);
    _Float16* x_h   = (_Float16*)take((size_t)TT * INP * 2);
    _Float16* x1_h  = (_Float16*)take((size_t)TT * BIc * 2);
    _Float16* hv_h  = (_Float16*)take((size_t)TT * BIc * 2);
    _Float16* lh_h  = (_Float16*)take((size_t)TT * MLP * 2);
    _Float16* hbuf  = (_Float16*)take((size_t)2 * 2 * HH * 2);
    float* XW   = (float*)take((size_t)2 * TT * G4 * 4);
    float* AhB  = (float*)take((size_t)TT * MLP * 4);
    float* AdB  = (float*)take((size_t)TT * MLP * 4);
    float* LhA  = (float*)take((size_t)TT * MLP * 4);
    float* cst  = (float*)take((size_t)2 * HH * 4);
    int*   cnt  = (int*)take((size_t)2 * 2 * TT * 4);

    auto cvt = [&](const float* s, int cols, _Float16* d, int dld, long rows) {
        long total = rows * (long)dld;
        convert2d<<<dim3((unsigned)((total + 255) / 256)), 256, 0, stream>>>(s, cols, d, dld, total);
    };
    auto gemm = [&](const _Float16* A, int lda, const int* ridx,
                    const _Float16* W, int ldw, const float* b1, const float* b2,
                    const float* Ci, float* Co, _Float16* CoH, int M, int N, int K, int act) {
        gemm_wmma<<<dim3((N + 16 * NT - 1) / (16 * NT), M / 16), 32, 0, stream>>>(
            A, lda, ridx, W, ldw, b1, b2, Ci, Co, CoH, M, N, K, act);
    };

    // 0) reset cooperative-scan counters (poisoned/stale otherwise)
    zero_i32<<<dim3((2 * 2 * TT + 255) / 256), 256, 0, stream>>>(cnt, 2 * 2 * TT);

    // 1) weights -> f16 (pad Wih0 K 400->416)
    cvt(Wih0, INQ, wih0h, INP, 2L * G4);
    cvt(Whh0, HH,  whh0h, HH,  2L * G4);
    cvt(Wih1, BIc, wih1h, BIc, 2L * G4);
    cvt(Whh1, HH,  whh1h, HH,  2L * G4);
    cvt(aW1,  MLP, aW1h,  MLP, MLP);
    cvt(lW1,  MLP, lW1h,  MLP, MLP);
    cvt(lW2,  MLP, lW2h,  MLP, LBL);

    // 2) embeddings -> x_h [T, INP] f16
    embed_concat<<<dim3((TT * INP + 255) / 256), 256, 0, stream>>>(words, tags, emb_w, emb_t, x_h);

    // 3) layer0: XW[d] = x @ Wih0[d]^T + bih0[d] + bhh0[d]   (WMMA)
    for (int d = 0; d < 2; ++d)
        gemm(x_h, INP, nullptr, wih0h + (size_t)d * G4 * INP, INP,
             bih0 + d * G4, bhh0 + d * G4, nullptr, XW + (size_t)d * TT * G4, nullptr,
             TT, G4, INP, 0);
    lstm_init<<<dim3((2 * HH + 255) / 256), 256, 0, stream>>>(h0, c0, 0, hbuf, cst);
    lstm_scan<<<dim3(GSCAN, 2), 256, 0, stream>>>(XW, whh0h, hbuf, cst, x1_h, cnt);

    // 4) layer1
    for (int d = 0; d < 2; ++d)
        gemm(x1_h, BIc, nullptr, wih1h + (size_t)d * G4 * BIc, BIc,
             bih1 + d * G4, bhh1 + d * G4, nullptr, XW + (size_t)d * TT * G4, nullptr,
             TT, G4, BIc, 0);
    lstm_init<<<dim3((2 * HH + 255) / 256), 256, 0, stream>>>(h0, c0, 1, hbuf, cst);
    lstm_scan<<<dim3(GSCAN, 2), 256, 0, stream>>>(XW, whh1h, hbuf, cst, hv_h, cnt + 2 * TT);

    // 5) arc MLP: Ah = hv @ aW1[:, :BI]^T ; Ad = hv @ aW1[:, BI:]^T  (WMMA)
    gemm(hv_h, BIc, nullptr, aW1h,       MLP, nullptr, nullptr, nullptr, AhB, nullptr, TT, MLP, BIc, 0);
    gemm(hv_h, BIc, nullptr, aW1h + BIc, MLP, nullptr, nullptr, nullptr, AdB, nullptr, TT, MLP, BIc, 0);
    pairwise_arc<<<dim3(TT / 16, TT / 16), dim3(16, 16), 0, stream>>>(AhB, AdB, ab1, aW2, ab2, out);

    // 6) label MLP: lh = tanh(hv@lW1a^T + hv[arcs]@lW1b^T + lb1); labels = lh@lW2^T + lb2
    gemm(hv_h, BIc, nullptr, lW1h,       MLP, nullptr, nullptr, nullptr, LhA, nullptr, TT, MLP, BIc, 0);
    gemm(hv_h, BIc, arcs,    lW1h + BIc, MLP, lb1, nullptr, LhA, nullptr, lh_h, TT, MLP, BIc, 1);
    gemm(lh_h, MLP, nullptr, lW2h, MLP, lb2, nullptr, nullptr, out + TT * TT, nullptr, TT, LBL, MLP, 0);
}
